// MultiBoxLoss_71708773974155
// MI455X (gfx1250) — compile-verified, matrix operand order checked by reading, not run on hardware
//
#include <hip/hip_runtime.h>
#include <hip/hip_bf16.h>
#include <math.h>

// ---- problem constants (match setup_inputs) ----
#define BATCH 64
#define NPRI  8732
#define NCLS  81
#define NOBJ  32
#define RATIO 3
#define THRESH 0.5f

typedef __attribute__((ext_vector_type(16))) _Float16 v16h;
typedef __attribute__((ext_vector_type(8)))  float    v8f;
// 16B-aligned float4 (priors / locs rows are naturally 16B aligned)
typedef float f4a __attribute__((ext_vector_type(4), aligned(16)));
// 4B-aligned float4 (score rows are 324B apart -> only dword aligned)
typedef float f4u __attribute__((ext_vector_type(4), aligned(4)));

// -------------------------------------------------------------------------
// Kernel 0: zero the loss accumulators (ws is poisoned 0xAA by the harness)
// -------------------------------------------------------------------------
__global__ void init_kernel(float* acc) {
    if (threadIdx.x == 0 && blockIdx.x == 0) { acc[0] = 0.0f; acc[1] = 0.0f; }
}

// -------------------------------------------------------------------------
// Kernel 1: per-batch prior<->object matching + SSD box encoding.
// One block per batch. Per-object best prior via packed u64 LDS atomic-max;
// scatter-overwrite applied in phase B. Priors/locs move as b128 vectors.
// -------------------------------------------------------------------------
__global__ __launch_bounds__(256)
void match_kernel(const float* __restrict__ boxes,   // [B,O,4] xy
                  const int*   __restrict__ labels,  // [B,O]
                  const float* __restrict__ priors,  // [P,4] cxcy
                  int*   __restrict__ tc,            // [B,P]
                  float* __restrict__ tl,            // [B,P,4]
                  float* __restrict__ ovmax_g,       // [B,P] scratch
                  int*   __restrict__ obj_g,         // [B,P] scratch
                  int*   __restrict__ num_pos)       // [B]
{
    __shared__ float sbox[NOBJ][4];
    __shared__ int   slab[NOBJ];
    __shared__ unsigned long long best[NOBJ];  // (f32bits(iou)<<32)|prior
    __shared__ int   sbp[NOBJ];
    __shared__ int   s_npos;

    const int b = blockIdx.x, tid = threadIdx.x;
    if (tid < NOBJ) {
        const f4a bv = *(const f4a*)(boxes + ((size_t)b*NOBJ + tid)*4);
        #pragma unroll
        for (int j = 0; j < 4; ++j) sbox[tid][j] = bv[j];
        slab[tid] = labels[b*NOBJ + tid];
        best[tid] = 0ull;
    }
    if (tid == 0) s_npos = 0;
    __syncthreads();

    unsigned long long lbest[NOBJ];
    #pragma unroll
    for (int o = 0; o < NOBJ; ++o) lbest[o] = 0ull;

    const size_t base = (size_t)b * NPRI;

    // ---- phase A: per-prior argmax over objects, per-object local best ----
    for (int p = tid; p < NPRI; p += blockDim.x) {
        const f4a pv = *(const f4a*)(priors + (size_t)p*4);   // global_load_b128
        const float pcx = pv[0], pcy = pv[1], pw = pv[2], ph = pv[3];
        const float px1 = pcx - 0.5f*pw, py1 = pcy - 0.5f*ph;
        const float px2 = pcx + 0.5f*pw, py2 = pcy + 0.5f*ph;
        const float areap = pw * ph;

        float bestov = -1.0f; int besto = 0;
        #pragma unroll
        for (int o = 0; o < NOBJ; ++o) {
            const float ax1 = sbox[o][0], ay1 = sbox[o][1];
            const float ax2 = sbox[o][2], ay2 = sbox[o][3];
            const float w = fmaxf(fminf(ax2, px2) - fmaxf(ax1, px1), 0.0f);
            const float h = fmaxf(fminf(ay2, py2) - fmaxf(ay1, py1), 0.0f);
            const float inter = w * h;
            const float areaa = (ax2 - ax1) * (ay2 - ay1);
            const float iou = inter / (areaa + areap - inter);
            if (iou > bestov) { bestov = iou; besto = o; }
            const unsigned long long pk =
                ((unsigned long long)__float_as_uint(iou) << 32) | (unsigned)p;
            if (pk > lbest[o]) lbest[o] = pk;
        }
        ovmax_g[base + p] = bestov;
        obj_g[base + p]   = besto;
    }
    #pragma unroll
    for (int o = 0; o < NOBJ; ++o) atomicMax(&best[o], lbest[o]);
    __syncthreads();

    if (tid < NOBJ) sbp[tid] = (int)(best[tid] & 0xFFFFFFFFull);
    __syncthreads();

    // ---- phase B: forced assignment, labels, box encoding ----
    int lpos = 0;
    for (int p = tid; p < NPRI; p += blockDim.x) {
        float ov = ovmax_g[base + p];
        int   oi = obj_g[base + p];
        #pragma unroll
        for (int o = 0; o < NOBJ; ++o)
            if (sbp[o] == p) { oi = o; ov = 1.0f; }   // last writer wins (scatter)

        int lab = slab[oi];
        if (ov < THRESH) lab = 0;
        tc[base + p] = lab;
        if (lab > 0) ++lpos;

        const float bx1 = sbox[oi][0], by1 = sbox[oi][1];
        const float bx2 = sbox[oi][2], by2 = sbox[oi][3];
        const float cx = 0.5f*(bx1+bx2), cy = 0.5f*(by1+by2);
        const float w  = bx2 - bx1,      h  = by2 - by1;
        const f4a pv = *(const f4a*)(priors + (size_t)p*4);
        f4a t;
        t[0] = (cx - pv[0]) / (pv[2] * 0.1f);
        t[1] = (cy - pv[1]) / (pv[3] * 0.1f);
        t[2] = __logf(w / pv[2]) * 5.0f;
        t[3] = __logf(h / pv[3]) * 5.0f;
        *(f4a*)(tl + (base + p)*4) = t;               // global_store_b128
    }
    atomicAdd(&s_npos, lpos);
    __syncthreads();
    if (tid == 0) num_pos[b] = s_npos;
}

// -------------------------------------------------------------------------
// Kernel 2: log-softmax (WMMA row-sum of exp), CE and mining key.
// One wave per 16-prior tile. Class partition is chosen for CONTIGUOUS
// per-lane loads: lane L (half 0) owns classes 0..47, lane L^16 owns
// 48..80 (+pads). WMMA chunk ch uses elements [16ch..16ch+16) of each
// half => each class slot appears exactly once across the 3 WMMAs, and
// with B = ones the f32 accumulator rows are exact exp row-sums
// regardless of K-slot permutation.
// Grid is exact: B*ceil(P/16)=34944 waves = 4368 blocks of 8 waves.
// -------------------------------------------------------------------------
__global__ __launch_bounds__(256)
void softmax_kernel(const float* __restrict__ scores,  // [B,P,C]
                    const int*   __restrict__ tc,      // [B,P]
                    float*       __restrict__ ce,      // [B,P]
                    unsigned*    __restrict__ key)     // [B,P]
{
    const int TP   = (NPRI + 15) / 16;            // 546 tiles per batch
    const int gwav = (blockIdx.x * 256 + threadIdx.x) >> 5;
    const int lane = threadIdx.x & 31;
    const int wv   = (threadIdx.x >> 5);

    const int b    = gwav / TP;
    const int tile = gwav % TP;
    const int row  = lane & 15;
    const int half = lane >> 4;
    const int p    = tile * 16 + row;
    const bool valid = (p < NPRI);

    // clamp the row pointer: tail rows read real (unused) data, no guards
    const size_t prow = (size_t)b * NPRI + (valid ? p : (NPRI - 1));
    const float* sp = scores + prow * NCLS;

    __builtin_prefetch(sp, 0, 1);                 // global_prefetch_b8
    __builtin_prefetch(sp + 64, 0, 1);

    // ---- pass 1: contiguous vector loads, per-lane max, pair-combine ----
    float vals[48];
    const float* lp = sp + half * 48;
    if (half == 0) {
        #pragma unroll
        for (int i = 0; i < 48; i += 4) {
            const f4u v = *(const f4u*)(lp + i);  // wide unconditional loads
            vals[i+0] = v[0]; vals[i+1] = v[1]; vals[i+2] = v[2]; vals[i+3] = v[3];
        }
    } else {
        #pragma unroll
        for (int i = 0; i < 32; i += 4) {
            const f4u v = *(const f4u*)(lp + i);  // classes 48..79
            vals[i+0] = v[0]; vals[i+1] = v[1]; vals[i+2] = v[2]; vals[i+3] = v[3];
        }
        vals[32] = lp[32];                        // class 80
        #pragma unroll
        for (int i = 33; i < 48; ++i) vals[i] = -1e30f;   // pads
    }
    float mx = -1e30f;
    #pragma unroll
    for (int i = 0; i < 48; ++i) mx = fmaxf(mx, vals[i]);
    mx = fmaxf(mx, __shfl_xor(mx, 16));

    // ---- pass 2: exp in f16, WMMA-with-ones accumulates row sums in f32 ----
    v16h bones;
    #pragma unroll
    for (int t = 0; t < 16; ++t) bones[t] = (_Float16)1.0f;

    v8f acc = {0.f, 0.f, 0.f, 0.f, 0.f, 0.f, 0.f, 0.f};
    #pragma unroll
    for (int ch = 0; ch < 3; ++ch) {
        v16h a;
        #pragma unroll
        for (int t = 0; t < 16; ++t) {
            const float e = __expf(vals[ch * 16 + t] - mx);   // pads underflow to 0
            a[t] = (_Float16)e;
        }
        acc = __builtin_amdgcn_wmma_f32_16x16x32_f16(
                  false, a, false, bones, (short)0, acc, false, false);
    }

    // D layout: lanes 0-15 hold rows 0..7 (VGPR r -> M=r), lanes 16-31 rows 8..15
    __shared__ float rowsum[8][16];
    if (lane == 0) {
        #pragma unroll
        for (int r = 0; r < 8; ++r) rowsum[wv][r] = acc[r];
    } else if (lane == 16) {
        #pragma unroll
        for (int r = 0; r < 8; ++r) rowsum[wv][8 + r] = acc[r];
    }
    asm volatile("s_wait_dscnt 0" ::: "memory");   // wave-local DS ordering (CDNA5 split counter)
    __syncthreads();

    const float rs     = rowsum[wv][row];
    const float logsum = mx + __logf(fmaxf(rs, 1e-30f));

    if (valid && half == 0) {
        const float s0  = sp[0];
        const int   cls = tc[(size_t)b * NPRI + p];
        const float stc = sp[cls];
        const float bg  = fmaxf(logsum - s0, 0.0f);    // strictly > 0 analytically
        ce[(size_t)b * NPRI + p]  = logsum - stc;
        // order-preserving u32 key; positives excluded with key 0
        key[(size_t)b * NPRI + p] = (cls > 0) ? 0u : (__float_as_uint(bg) + 1u);
    }
}

// -------------------------------------------------------------------------
// Kernel 3: hard-negative mining (exact k-th-largest via 32-step binary
// search on key bits, row resident in dynamic LDS) + masked CE sum +
// smooth-L1 over positives (b128 loc loads). One block per batch.
// -------------------------------------------------------------------------
__global__ __launch_bounds__(512)
void mine_kernel(const float*    __restrict__ plocs,   // [B,P,4]
                 const float*    __restrict__ tl,      // [B,P,4]
                 const int*      __restrict__ tc,      // [B,P]
                 const float*    __restrict__ ce_g,    // [B,P]
                 const unsigned* __restrict__ key_g,   // [B,P]
                 const int*      __restrict__ num_pos, // [B]
                 float*          __restrict__ acc)     // [2]
{
    extern __shared__ unsigned char dynsm[];
    unsigned* keys = (unsigned*)dynsm;                     // P u32
    float*    ces  = (float*)(dynsm + sizeof(unsigned) * NPRI);

    __shared__ int   s_cnt;
    __shared__ float sc, sl;

    const int b = blockIdx.x, tid = threadIdx.x, nt = blockDim.x;
    const size_t base = (size_t)b * NPRI;

    for (int p = tid; p < NPRI; p += nt) {
        keys[p] = key_g[base + p];
        ces[p]  = ce_g[base + p];
    }
    if (tid == 0) { sc = 0.f; sl = 0.f; }
    __syncthreads();

    const int k = RATIO * num_pos[b];

    // largest t with count(keys >= t) >= k  (keys distinct w.h.p.; pos have key 0)
    unsigned lo = 1u, hi = 0xFFFFFFFFu;
    while (lo < hi) {
        const unsigned mid = lo + (hi - lo + 1u) / 2u;
        __syncthreads();
        if (tid == 0) s_cnt = 0;
        __syncthreads();
        int c = 0;
        for (int p = tid; p < NPRI; p += nt) if (keys[p] >= mid) ++c;
        atomicAdd(&s_cnt, c);
        __syncthreads();
        if (s_cnt >= k) lo = mid; else hi = mid - 1u;
    }
    const unsigned tstar = lo;

    float csum = 0.f, lsum = 0.f;
    for (int p = tid; p < NPRI; p += nt) {
        const int  cls = tc[base + p];
        const bool pos = (cls > 0);
        const bool sel = pos || (k > 0 && keys[p] >= tstar);
        if (sel) csum += ces[p];
        if (pos) {
            const f4a pl = *(const f4a*)(plocs + (base + p)*4);  // b128
            const f4a tt = *(const f4a*)(tl    + (base + p)*4);  // b128
            #pragma unroll
            for (int j = 0; j < 4; ++j) {
                const float d = fabsf(pl[j] - tt[j]);
                lsum += (d < 1.0f) ? 0.5f * d * d : d - 0.5f;
            }
        }
    }
    atomicAdd(&sc, csum);
    atomicAdd(&sl, lsum);
    __syncthreads();
    if (tid == 0) {
        atomicAdd(&acc[0], sc);
        atomicAdd(&acc[1], sl);
    }
}

// -------------------------------------------------------------------------
// Kernel 4: divide by total positives -> two scalar outputs
// -------------------------------------------------------------------------
__global__ void finalize_kernel(const float* __restrict__ acc,
                                const int*   __restrict__ num_pos,
                                float*       __restrict__ out)
{
    if (threadIdx.x == 0 && blockIdx.x == 0) {
        int n = 0;
        for (int b = 0; b < BATCH; ++b) n += num_pos[b];
        const float fn = (float)(n > 0 ? n : 1);
        out[0] = acc[0] / fn;
        out[1] = acc[1] / fn;
    }
}

// -------------------------------------------------------------------------
extern "C" void kernel_launch(void* const* d_in, const int* in_sizes, int n_in,
                              void* d_out, int out_size, void* d_ws, size_t ws_size,
                              hipStream_t stream)
{
    const float* plocs  = (const float*)d_in[0];   // [B,P,4]
    const float* scores = (const float*)d_in[1];   // [B,P,C]
    const float* boxes  = (const float*)d_in[2];   // [B,O,4]
    const int*   labels = (const int*)  d_in[3];   // [B,O]
    const float* priors = (const float*)d_in[4];   // [P,4]
    float* out = (float*)d_out;

    // workspace layout (bytes); all region offsets are 16B-aligned
    const size_t BP = (size_t)BATCH * NPRI;
    unsigned char* ws = (unsigned char*)d_ws;
    int*      tc      = (int*)      (ws);                         // BP ints
    float*    tl      = (float*)    (ws + BP * 4);                // BP*4 floats
    float*    ce      = (float*)    (ws + BP * 4  + BP * 16);     // BP floats
    unsigned* key     = (unsigned*) (ws + BP * 8  + BP * 16);     // BP u32
    float*    ovmax   = (float*)    (ws + BP * 12 + BP * 16);     // BP floats
    int*      obj     = (int*)      (ws + BP * 16 + BP * 16);     // BP ints
    int*      np      = (int*)      (ws + BP * 20 + BP * 16);     // BATCH ints
    float*    acc     = (float*)    (ws + BP * 20 + BP * 16 + BATCH * 4); // 2 floats

    init_kernel<<<1, 1, 0, stream>>>(acc);

    match_kernel<<<BATCH, 256, 0, stream>>>(boxes, labels, priors,
                                            tc, tl, ovmax, obj, np);

    const int TP = (NPRI + 15) / 16;                  // 546
    const int nwaves = BATCH * TP;                    // 34944
    const int nblocks = nwaves / 8;                   // 4368 (exact, 8 waves/block)
    softmax_kernel<<<nblocks, 256, 0, stream>>>(scores, tc, ce, key);

    const size_t dyn = (size_t)NPRI * (sizeof(unsigned) + sizeof(float)); // ~68 KB < 320 KB/WGP
    mine_kernel<<<BATCH, 512, dyn, stream>>>(plocs, tl, tc, ce, key, np, acc);

    finalize_kernel<<<1, 1, 0, stream>>>(acc, np, out);
}
